// SoftAttentionDecoderRNN_35820027249357
// MI455X (gfx1250) — matrix operand order repaired and checked by reading, not compiled, not run on hardware
//
#include <hip/hip_runtime.h>
#include <cstdint>
#include <cstddef>

// Problem dims (fixed by the reference)
#define B_  64
#define L_  49
#define C_  2048
#define T_  20
#define H_  512
#define DE_ 256
#define V_  32000
#define KX_ (DE_ + C_ + H_)   // 2816 : [emb | gate*ctx | h] combined K for LSTM gates

typedef __attribute__((ext_vector_type(16))) __bf16 v16bf;
typedef __attribute__((ext_vector_type(8)))  float  v8f;

// ---------- helpers ----------
__device__ __forceinline__ unsigned short f2bf(float f) {
  union { float f; unsigned int u; } x; x.f = f;
  unsigned int u = x.u;
  unsigned int r = (u + 0x7FFFu + ((u >> 16) & 1u)) >> 16;  // round-to-nearest-even
  return (unsigned short)r;
}
__device__ __forceinline__ float sigmoidf_(float x) { return 1.0f / (1.0f + __expf(-x)); }

// ---------- generic bf16 WMMA GEMM (workgroup-tiled, async B staging) ----------
// C[M,N] = A[M,K](bf16,row-major,lda) @ Bt[N,K](bf16,row-major)^T + bias0[n] + bias1[n]
// Block = 128 threads = 4 waves, computes a 64(M) x 64(N) tile.
// Per K-step (32): the 64x32 bf16 weight block (4 KB) is staged into LDS ONCE per
// block via CDNA5 async global->LDS loads (ASYNCcnt), then each wave reads its B
// fragments from LDS (ds_load_b128) and owns one 16-row M-tile (4 WMMAs/step).
// grid = (N/64, M/64). N%64==0, M%64==0, K%32==0 required.
__global__ __launch_bounds__(128) void k_gemm_bf16_wmma(
    const unsigned short* __restrict__ A, int lda,
    const unsigned short* __restrict__ Bt, int K,
    const float* __restrict__ bias0, const float* __restrict__ bias1,
    float* __restrict__ Cout, long long ldc)
{
  __shared__ unsigned short smemB[64 * 32];   // 64 N-rows x 32 K  (4 KB)

  const int tid  = threadIdx.x;
  const int lane = tid & 31;
  const int wv   = tid >> 5;        // wave id 0..3 -> M sub-tile
  const int mrow = lane & 15;
  const int half = lane >> 4;
  const int tileM0 = blockIdx.y * 64;
  const int n0 = blockIdx.x * 64;

  // A row for this wave's M tile (row m = tileM0 + wv*16 + mrow)
  const unsigned short* Arow = A + (size_t)(tileM0 + wv * 16 + mrow) * (size_t)lda;

  // Async-staging assignment: 256 16-byte chunks (64 rows x 4 segs), 2 per thread.
  // chunk c: row = c>>2 (N row within block), seg = c&3 (16B segment of the 64B row)
  const unsigned lds_base = (unsigned)(uintptr_t)smemB;  // LDS byte offset (addr[31:0])
  const int ch0 = tid;            // chunks tid and tid+128
  const int r0 = ch0 >> 2,        s0 = ch0 & 3;
  const int r1 = (ch0 + 128) >> 2, s1 = (ch0 + 128) & 3;
  const unsigned short* gB0 = Bt + (size_t)(n0 + r0) * (size_t)K + s0 * 8;
  const unsigned short* gB1 = Bt + (size_t)(n0 + r1) * (size_t)K + s1 * 8;
  const unsigned ldsOff0 = lds_base + (unsigned)(r0 * 64 + s0 * 16);
  const unsigned ldsOff1 = lds_base + (unsigned)(r1 * 64 + s1 * 16);

  // LDS read pointers for this lane's B fragments: tile j -> row j*16 + (lane&15),
  // byte offset half*32 (+16) giving K = half*16 + 0..7 / 8..15.
  const char* smemBytes = (const char*)smemB;
  const char* bRd = smemBytes + (size_t)(mrow * 64 + half * 32);

  v8f acc0 = {}; v8f acc1 = {}; v8f acc2 = {}; v8f acc3 = {};

  for (int k = 0; k < K; k += 32) {
    // ---- async stage B block (k..k+31) into LDS ----
    asm volatile("global_load_async_to_lds_b128 %0, %1, off"
                 :: "v"(ldsOff0), "v"(gB0 + k) : "memory");
    asm volatile("global_load_async_to_lds_b128 %0, %1, off"
                 :: "v"(ldsOff1), "v"(gB1 + k) : "memory");

    // ---- A fragment from global (16x32 bf16, contiguous K = half*16 + 0..15, same order as B) ----
    union { v16bf v; uint4 u[2]; } a;
    a.u[0] = *(const uint4*)(Arow + k + half * 16);
    a.u[1] = *(const uint4*)(Arow + k + half * 16 + 8);

    asm volatile("s_wait_asynccnt 0" ::: "memory");
    __syncthreads();

    // ---- B fragments from LDS, 4 WMMAs (this wave's M tile x 4 N tiles) ----
    union { v16bf v; uint4 u[2]; } b0, b1, b2, b3;
    b0.u[0] = *(const uint4*)(bRd +    0); b0.u[1] = *(const uint4*)(bRd +    0 + 16);
    b1.u[0] = *(const uint4*)(bRd + 1024); b1.u[1] = *(const uint4*)(bRd + 1024 + 16);
    b2.u[0] = *(const uint4*)(bRd + 2048); b2.u[1] = *(const uint4*)(bRd + 2048 + 16);
    b3.u[0] = *(const uint4*)(bRd + 3072); b3.u[1] = *(const uint4*)(bRd + 3072 + 16);

    acc0 = __builtin_amdgcn_wmma_f32_16x16x32_bf16(false, a.v, false, b0.v, (short)0, acc0, false, false);
    acc1 = __builtin_amdgcn_wmma_f32_16x16x32_bf16(false, a.v, false, b1.v, (short)0, acc1, false, false);
    acc2 = __builtin_amdgcn_wmma_f32_16x16x32_bf16(false, a.v, false, b2.v, (short)0, acc2, false, false);
    acc3 = __builtin_amdgcn_wmma_f32_16x16x32_bf16(false, a.v, false, b3.v, (short)0, acc3, false, false);

    __syncthreads();   // protect LDS before next iteration's staging overwrites it
  }

  // C/D layout: VGPR r, lanes 0-15 -> (M=r, N=lane), lanes 16-31 -> (M=8+r, N=lane-16)
  const int nn = lane & 15;
  const int mo = (lane >> 4) * 8;
  float bv0 = 0.f, bv1 = 0.f, bv2 = 0.f, bv3 = 0.f;
  if (bias0) { bv0 += bias0[n0 + nn]; bv1 += bias0[n0 + 16 + nn]; bv2 += bias0[n0 + 32 + nn]; bv3 += bias0[n0 + 48 + nn]; }
  if (bias1) { bv0 += bias1[n0 + nn]; bv1 += bias1[n0 + 16 + nn]; bv2 += bias1[n0 + 32 + nn]; bv3 += bias1[n0 + 48 + nn]; }
#pragma unroll
  for (int r = 0; r < 8; ++r) {
    const int m = tileM0 + wv * 16 + mo + r;
    float* crow = Cout + (size_t)m * (size_t)ldc;
    crow[n0 +      nn] = acc0[r] + bv0;
    crow[n0 + 16 + nn] = acc1[r] + bv1;
    crow[n0 + 32 + nn] = acc2[r] + bv2;
    crow[n0 + 48 + nn] = acc3[r] + bv3;
  }
}

// ---------- weight prep ----------
// dst[N,K](bf16) = transpose(src[K,N](f32)) ; grid = (N, ceil(K/256)), block 256
__global__ void k_transpose_bf16(unsigned short* __restrict__ dst,
                                 const float* __restrict__ src, int K, int N) {
  const int n = blockIdx.x;
  const int k = blockIdx.y * 256 + threadIdx.x;
  if (k < K) dst[(size_t)n * K + k] = f2bf(src[(size_t)k * N + n]);
}

// dst[r, 0..cols) (bf16, row stride ldd) = src[r, 0..cols) (f32, row stride lds)
__global__ void k_convert_rows(unsigned short* __restrict__ dst, int ldd,
                               const float* __restrict__ src, int lds, int cols) {
  const int r = blockIdx.x;
  const int c = blockIdx.y * 256 + threadIdx.x;
  if (c < cols) dst[(size_t)r * ldd + c] = f2bf(src[(size_t)r * lds + c]);
}

// fmean_bf[b,c] = bf16(mean_l features[b,l,c]) ; grid=(B, C/256), block 256
__global__ void k_fmean(unsigned short* __restrict__ fmean_bf,
                        const float* __restrict__ features) {
  const int b = blockIdx.x;
  const int c = blockIdx.y * 256 + threadIdx.x;
  float s = 0.f;
  for (int l = 0; l < L_; ++l) s += features[((size_t)b * L_ + l) * C_ + c];
  fmean_bf[(size_t)b * C_ + c] = f2bf(s * (1.0f / (float)L_));
}

// h_bf = bf16(h); xh[:,2304:2816] = bf16(h) ; grid=B, block=H
__global__ void k_h_to_bf(unsigned short* __restrict__ h_bf,
                          unsigned short* __restrict__ xh,
                          const float* __restrict__ h) {
  const int b = blockIdx.x, j = threadIdx.x;
  const unsigned short v = f2bf(h[(size_t)b * H_ + j]);
  h_bf[(size_t)b * H_ + j] = v;
  xh[(size_t)b * KX_ + (DE_ + C_) + j] = v;
}

// e[b,l] = sum_h relu(att_img[b,l,h] + att_h[b,h]) * w_comb[h] + b_comb
__global__ void k_attn_e(const float* __restrict__ att_img,
                         const float* __restrict__ att_h,
                         const float* __restrict__ w_comb,
                         const float* __restrict__ b_comb,
                         float* __restrict__ e) {
  __shared__ float red[256];
  const int idx = blockIdx.x;
  const int b = idx / L_;
  const float* ai = att_img + (size_t)idx * H_;
  const float* ah = att_h + (size_t)b * H_;
  float s = 0.f;
  for (int h = threadIdx.x; h < H_; h += 256) {
    float v = ai[h] + ah[h];
    v = v > 0.f ? v : 0.f;
    s += v * w_comb[h];
  }
  red[threadIdx.x] = s; __syncthreads();
  for (int off = 128; off > 0; off >>= 1) {
    if (threadIdx.x < off) red[threadIdx.x] += red[threadIdx.x + off];
    __syncthreads();
  }
  if (threadIdx.x == 0) e[idx] = red[0] + b_comb[0];
}

// in-place softmax over L per batch row ; grid=B, block=32
__global__ void k_softmax(float* __restrict__ e) {
  if (threadIdx.x != 0) return;
  float* row = e + (size_t)blockIdx.x * L_;
  float mx = -3.0e38f;
  for (int l = 0; l < L_; ++l) mx = fmaxf(mx, row[l]);
  float s = 0.f, tmp[L_];
  for (int l = 0; l < L_; ++l) { tmp[l] = __expf(row[l] - mx); s += tmp[l]; }
  const float inv = 1.0f / s;
  for (int l = 0; l < L_; ++l) row[l] = tmp[l] * inv;
}

// ctx[b,c] = sum_l alphas[b,l] * features[b,l,c] ; grid=(B, C/256), block 256
__global__ void k_ctx(float* __restrict__ ctx, const float* __restrict__ alphas,
                      const float* __restrict__ features) {
  const int b = blockIdx.x;
  const int c = blockIdx.y * 256 + threadIdx.x;
  float s = 0.f;
  for (int l = 0; l < L_; ++l)
    s += alphas[(size_t)b * L_ + l] * features[((size_t)b * L_ + l) * C_ + c];
  ctx[(size_t)b * C_ + c] = s;
}

// xh[:,0:256] = bf16(embed[captions[b,t]]), xh[:,256:2304] = bf16(sigmoid(graw)*ctx)
__global__ void k_gate_x(unsigned short* __restrict__ xh,
                         const int* __restrict__ captions,
                         const float* __restrict__ embed,
                         const float* __restrict__ graw,
                         const float* __restrict__ ctx, int t) {
  const int b = blockIdx.x;
  const int cap = captions[(size_t)b * T_ + t];
  unsigned short* row = xh + (size_t)b * KX_;
  { // embedding slice
    const int j = threadIdx.x;
    row[j] = f2bf(embed[(size_t)cap * DE_ + j]);
  }
  for (int c = threadIdx.x; c < C_; c += 256) {
    const float g = sigmoidf_(graw[(size_t)b * C_ + c]);
    row[DE_ + c] = f2bf(g * ctx[(size_t)b * C_ + c]);
  }
}

// torch LSTM cell elementwise; updates c, h, h_bf, xh tail ; grid=B, block=H
__global__ void k_lstm_cell(const float* __restrict__ gates,
                            float* __restrict__ c, float* __restrict__ h,
                            unsigned short* __restrict__ h_bf,
                            unsigned short* __restrict__ xh) {
  const int b = blockIdx.x, j = threadIdx.x;
  const float* g = gates + (size_t)b * (4 * H_);
  const float gi = g[j], gf = g[H_ + j], gg = g[2 * H_ + j], go = g[3 * H_ + j];
  const float cn = sigmoidf_(gf) * c[(size_t)b * H_ + j] + sigmoidf_(gi) * tanhf(gg);
  const float hn = sigmoidf_(go) * tanhf(cn);
  c[(size_t)b * H_ + j] = cn;
  h[(size_t)b * H_ + j] = hn;
  const unsigned short hb = f2bf(hn);
  h_bf[(size_t)b * H_ + j] = hb;
  xh[(size_t)b * KX_ + (DE_ + C_) + j] = hb;
}

// ---------- launch ----------
extern "C" void kernel_launch(void* const* d_in, const int* in_sizes, int n_in,
                              void* d_out, int out_size, void* d_ws, size_t ws_size,
                              hipStream_t stream) {
  (void)in_sizes; (void)n_in; (void)out_size; (void)ws_size;
  const float* features = (const float*)d_in[0];
  const int*   captions = (const int*)d_in[1];
  const float* embed    = (const float*)d_in[2];
  const float* W_img    = (const float*)d_in[3];
  const float* b_img    = (const float*)d_in[4];
  const float* W_lh     = (const float*)d_in[5];
  const float* b_lh     = (const float*)d_in[6];
  const float* w_comb   = (const float*)d_in[7];
  const float* b_comb   = (const float*)d_in[8];
  const float* W_init_h = (const float*)d_in[9];
  const float* b_init_h = (const float*)d_in[10];
  const float* W_init_c = (const float*)d_in[11];
  const float* b_init_c = (const float*)d_in[12];
  const float* W_beta   = (const float*)d_in[13];
  const float* b_beta   = (const float*)d_in[14];
  const float* W_ih     = (const float*)d_in[15];
  const float* b_ih     = (const float*)d_in[16];
  const float* W_hh     = (const float*)d_in[17];
  const float* b_hh     = (const float*)d_in[18];
  const float* W_out    = (const float*)d_in[19];
  const float* b_out    = (const float*)d_in[20];
  float* out = (float*)d_out;
  char* ws = (char*)d_ws;

  // workspace layout (all 256B-aligned)
  size_t o = 0;
  unsigned short* WOUT_T  = (unsigned short*)(ws + o); o += (size_t)V_ * H_ * 2;          // 32.77 MB
  unsigned short* WCOMB   = (unsigned short*)(ws + o); o += (size_t)(4 * H_) * KX_ * 2;   // 11.53 MB
  unsigned short* WIMG_T  = (unsigned short*)(ws + o); o += (size_t)H_ * C_ * 2;
  unsigned short* WINH_T  = (unsigned short*)(ws + o); o += (size_t)H_ * C_ * 2;
  unsigned short* WINC_T  = (unsigned short*)(ws + o); o += (size_t)H_ * C_ * 2;
  unsigned short* WLH_T   = (unsigned short*)(ws + o); o += (size_t)H_ * H_ * 2;
  unsigned short* WBETA_T = (unsigned short*)(ws + o); o += (size_t)C_ * H_ * 2;
  unsigned short* FEATBF  = (unsigned short*)(ws + o); o += (size_t)B_ * L_ * C_ * 2;     // 12.85 MB
  unsigned short* FMEANBF = (unsigned short*)(ws + o); o += (size_t)B_ * C_ * 2;
  float* ATTIMG  = (float*)(ws + o); o += (size_t)B_ * L_ * H_ * 4;                        // 6.42 MB
  float* Hbuf    = (float*)(ws + o); o += (size_t)B_ * H_ * 4;
  float* Cbuf    = (float*)(ws + o); o += (size_t)B_ * H_ * 4;
  unsigned short* HBF = (unsigned short*)(ws + o); o += (size_t)B_ * H_ * 2;
  float* ATTH    = (float*)(ws + o); o += (size_t)B_ * H_ * 4;
  float* EBUF    = (float*)(ws + o); o += 12800;                                          // B*L padded
  float* CTX     = (float*)(ws + o); o += (size_t)B_ * C_ * 4;
  float* GRAW    = (float*)(ws + o); o += (size_t)B_ * C_ * 4;
  unsigned short* XH = (unsigned short*)(ws + o); o += (size_t)B_ * KX_ * 2;
  float* GATES   = (float*)(ws + o); o += (size_t)B_ * 4 * H_ * 4;

  // ---- one-time weight prep (per launch) ----
  k_transpose_bf16<<<dim3(V_, 2), 256, 0, stream>>>(WOUT_T, W_out, H_, V_);        // (H,V)->(V,H)
  k_transpose_bf16<<<dim3(H_, 8), 256, 0, stream>>>(WIMG_T, W_img, C_, H_);        // (C,H)->(H,C)
  k_transpose_bf16<<<dim3(H_, 8), 256, 0, stream>>>(WINH_T, W_init_h, C_, H_);
  k_transpose_bf16<<<dim3(H_, 8), 256, 0, stream>>>(WINC_T, W_init_c, C_, H_);
  k_transpose_bf16<<<dim3(H_, 2), 256, 0, stream>>>(WLH_T, W_lh, H_, H_);          // (H,H)->(H,H)^T
  k_transpose_bf16<<<dim3(C_, 2), 256, 0, stream>>>(WBETA_T, W_beta, H_, C_);      // (H,C)->(C,H)
  // W_ih (2048 x 2304) and W_hh (2048 x 512) are already (N,K): concat along K.
  k_convert_rows<<<dim3(4 * H_, (DE_ + C_) / 256), 256, 0, stream>>>(WCOMB, KX_, W_ih, DE_ + C_, DE_ + C_);
  k_convert_rows<<<dim3(4 * H_, H_ / 256), 256, 0, stream>>>(WCOMB + (DE_ + C_), KX_, W_hh, H_, H_);
  // features -> bf16 A matrix (3136 x 2048)
  k_convert_rows<<<dim3(B_ * L_, C_ / 256), 256, 0, stream>>>(FEATBF, C_, features, C_, C_);
  k_fmean<<<dim3(B_, C_ / 256), 256, 0, stream>>>(FMEANBF, features);

  // ---- h0 / c0 : (64 x 2048) @ (2048 x 512) ----
  k_gemm_bf16_wmma<<<dim3(H_ / 64, 1), 128, 0, stream>>>(FMEANBF, C_, WINH_T, C_, b_init_h, nullptr, Hbuf, H_);
  k_gemm_bf16_wmma<<<dim3(H_ / 64, 1), 128, 0, stream>>>(FMEANBF, C_, WINC_T, C_, b_init_c, nullptr, Cbuf, H_);
  k_h_to_bf<<<B_, H_, 0, stream>>>(HBF, XH, Hbuf);

  // ---- att_img : (3136 x 2048) @ (2048 x 512) + b_img ----
  k_gemm_bf16_wmma<<<dim3(H_ / 64, (B_ * L_) / 64), 128, 0, stream>>>(FEATBF, C_, WIMG_T, C_, b_img, nullptr, ATTIMG, H_);

  // ---- recurrence ----
  for (int t = 0; t < T_; ++t) {
    // att_h = h @ W_lh + b_lh : (64x512)@(512x512)
    k_gemm_bf16_wmma<<<dim3(H_ / 64, 1), 128, 0, stream>>>(HBF, H_, WLH_T, H_, b_lh, nullptr, ATTH, H_);
    // attention scores, softmax, context
    k_attn_e<<<B_ * L_, 256, 0, stream>>>(ATTIMG, ATTH, w_comb, b_comb, EBUF);
    k_softmax<<<B_, 32, 0, stream>>>(EBUF);
    k_ctx<<<dim3(B_, C_ / 256), 256, 0, stream>>>(CTX, EBUF, features);
    // gate = sigmoid(h @ W_beta + b_beta) : (64x512)@(512x2048)
    k_gemm_bf16_wmma<<<dim3(C_ / 64, 1), 128, 0, stream>>>(HBF, H_, WBETA_T, H_, b_beta, nullptr, GRAW, C_);
    // x = [emb_t, gate*ctx] (h tail already in XH)
    k_gate_x<<<B_, 256, 0, stream>>>(XH, captions, embed, GRAW, CTX, t);
    // gates = [x|h] @ [W_ih|W_hh]^T + b_ih + b_hh : (64x2816)@(2816x2048)
    k_gemm_bf16_wmma<<<dim3((4 * H_) / 64, 1), 128, 0, stream>>>(XH, KX_, WCOMB, KX_, b_ih, b_hh, GATES, 4 * H_);
    // LSTM cell elementwise; refresh h/c, h_bf, xh tail
    k_lstm_cell<<<B_, H_, 0, stream>>>(GATES, Cbuf, Hbuf, HBF, XH);
    // logits = h_new @ W_out + b_out : (64x512)@(512x32000) -> out[:, t, :]
    k_gemm_bf16_wmma<<<dim3(V_ / 64, 1), 128, 0, stream>>>(HBF, H_, WOUT_T, H_, b_out, nullptr,
                                                           out + (size_t)t * V_, (long long)T_ * V_);
  }
}
